// DeepseekV4SharedExpert_67637144978441
// MI455X (gfx1250) — 8-bit matrix operand order repaired and checked by reading, compiled, not run on hardware
//
#include <hip/hip_runtime.h>
#include <hip/hip_bf16.h>

typedef __attribute__((ext_vector_type(8))) int v8i;

#define H_DIM 4096
#define I_DIM 2048
#define QMAXF 127.0f
#define SWIGLU_LIMIT 10.0f
#define KSTEP 128
#define LDS_STRIDE 144

__device__ __forceinline__ v8i wmma_iu8(v8i a, v8i b, v8i c) {
    return __builtin_amdgcn_wmma_i32_16x16x64_iu8(true, a, true, b, c, false, false);
}

__device__ __forceinline__ unsigned lds_off32(const void* p) {
    return (unsigned)(unsigned long long)(uintptr_t)p;
}
__device__ __forceinline__ void async_b128(unsigned lds_off, const signed char* g) {
    asm volatile("global_load_async_to_lds_b128 %0, %1, off"
                 :: "v"(lds_off), "v"((unsigned long long)(uintptr_t)g)
                 : "memory");
}

__device__ __forceinline__ v8i lds_a_frag(const signed char* tile, int rowLoc,
                                          int ksub, int lane) {
    const signed char* p = tile + (rowLoc + (lane & 15)) * LDS_STRIDE + ksub
                                + ((lane >> 4) << 4);
    int4 d0 = *(const int4*)(p);
    int4 d1 = *(const int4*)(p + 32);
    v8i f = {d0.x, d0.y, d0.z, d0.w, d1.x, d1.y, d1.z, d1.w};
    return f;
}
__device__ __forceinline__ v8i lds_b_frag(const signed char* tile, int colLoc,
                                          int ksub, int lane) {
    const signed char* p = tile + (colLoc + (lane & 15)) * LDS_STRIDE + ksub
                                + ((lane >> 4) << 4);
    int4 d0 = *(const int4*)(p);
    int4 d1 = *(const int4*)(p + 32);
    v8i f = {d0.x, d0.y, d0.z, d0.w, d1.x, d1.y, d1.z, d1.w};
    return f;
}

__global__ __launch_bounds__(256) void quant_x_kernel(
    const float* __restrict__ x,
    const float* __restrict__ inv_g, const float* __restrict__ inv_u,
    signed char* __restrict__ qg, signed char* __restrict__ qu,
    float* __restrict__ sg, float* __restrict__ su)
{
    __shared__ float red_g[256];
    __shared__ float red_u[256];
    const int t = blockIdx.x;
    const float* xr = x + (size_t)t * H_DIM;

    float vg[16], vu[16];
    float mg = 0.0f, mu = 0.0f;
#pragma unroll
    for (int i = 0; i < 16; ++i) {
        int d = threadIdx.x + i * 256;
        float xb = __bfloat162float(__float2bfloat16(xr[d]));
        float g = xb * inv_g[d];
        float u = xb * inv_u[d];
        vg[i] = g; vu[i] = u;
        mg = fmaxf(mg, fabsf(g));
        mu = fmaxf(mu, fabsf(u));
    }
    red_g[threadIdx.x] = mg;
    red_u[threadIdx.x] = mu;
    __syncthreads();
    for (int s = 128; s > 0; s >>= 1) {
        if (threadIdx.x < s) {
            red_g[threadIdx.x] = fmaxf(red_g[threadIdx.x], red_g[threadIdx.x + s]);
            red_u[threadIdx.x] = fmaxf(red_u[threadIdx.x], red_u[threadIdx.x + s]);
        }
        __syncthreads();
    }
    float scg = fmaxf(red_g[0] * (1.0f / QMAXF), 1e-8f);
    float scu = fmaxf(red_u[0] * (1.0f / QMAXF), 1e-8f);
    if (threadIdx.x == 0) { sg[t] = scg; su[t] = scu; }
    float rg = 1.0f / scg, ru = 1.0f / scu;
#pragma unroll
    for (int i = 0; i < 16; ++i) {
        int d = threadIdx.x + i * 256;
        qg[(size_t)t * H_DIM + d] =
            (signed char)(int)fminf(fmaxf(rintf(vg[i] * rg), -QMAXF), QMAXF);
        qu[(size_t)t * H_DIM + d] =
            (signed char)(int)fminf(fmaxf(rintf(vu[i] * ru), -QMAXF), QMAXF);
    }
}

__global__ __launch_bounds__(256) void convert_w_kernel(
    const float* __restrict__ w, signed char* __restrict__ wq, int n)
{
    int i = blockIdx.x * 256 + threadIdx.x;
    if (i < n) wq[i] = (signed char)__float2int_rn(w[i]);
}

__global__ __launch_bounds__(256) void gemm_gate_up_kernel(
    const signed char* __restrict__ qg, const signed char* __restrict__ qu,
    const float* __restrict__ sg, const float* __restrict__ su,
    const signed char* __restrict__ wg, const signed char* __restrict__ wu,
    const float* __restrict__ swg, const float* __restrict__ swu,
    __hip_bfloat16* __restrict__ inter)
{
    __shared__ signed char sAg[2][64 * LDS_STRIDE];
    __shared__ signed char sAu[2][64 * LDS_STRIDE];
    __shared__ signed char sBg[2][128 * LDS_STRIDE];
    __shared__ signed char sBu[2][128 * LDS_STRIDE];

    const int tid  = threadIdx.x;
    const int lane = tid & 31;
    const int wave = tid >> 5;
    const int rowB = blockIdx.y * 64;
    const int colB = blockIdx.x * 128;
    const int rowW = (wave & 1) * 32;
    const int colW = (wave >> 1) * 32;

    auto stage = [&](int buf, int k0) {
#pragma unroll
        for (int j = 0; j < 2; ++j) {
            int c = tid + (j << 8);
            int r = c >> 3, o = (c & 7) << 4;
            async_b128(lds_off32(&sAg[buf][r * LDS_STRIDE + o]),
                       qg + (size_t)(rowB + r) * H_DIM + k0 + o);
            async_b128(lds_off32(&sAu[buf][r * LDS_STRIDE + o]),
                       qu + (size_t)(rowB + r) * H_DIM + k0 + o);
        }
#pragma unroll
        for (int j = 0; j < 4; ++j) {
            int c = tid + (j << 8);
            int r = c >> 3, o = (c & 7) << 4;
            async_b128(lds_off32(&sBg[buf][r * LDS_STRIDE + o]),
                       wg + (size_t)(colB + r) * H_DIM + k0 + o);
            async_b128(lds_off32(&sBu[buf][r * LDS_STRIDE + o]),
                       wu + (size_t)(colB + r) * H_DIM + k0 + o);
        }
    };

    v8i accG[2][2], accU[2][2];
#pragma unroll
    for (int mi = 0; mi < 2; ++mi)
#pragma unroll
        for (int ni = 0; ni < 2; ++ni) {
            accG[mi][ni] = (v8i)(0);
            accU[mi][ni] = (v8i)(0);
        }

    stage(0, 0);
    for (int k0 = 0; k0 < H_DIM; k0 += KSTEP) {
        const int buf = (k0 >> 7) & 1;
        __syncthreads();
        if (k0 + KSTEP < H_DIM) {
            stage(buf ^ 1, k0 + KSTEP);
            asm volatile("s_wait_asynccnt 0xc" ::: "memory");
        } else {
            asm volatile("s_wait_asynccnt 0x0" ::: "memory");
        }
        __syncthreads();

#pragma unroll
        for (int ks = 0; ks < KSTEP; ks += 64) {
            v8i aG0 = lds_a_frag(sAg[buf], rowW,      ks, lane);
            v8i aG1 = lds_a_frag(sAg[buf], rowW + 16, ks, lane);
            v8i aU0 = lds_a_frag(sAu[buf], rowW,      ks, lane);
            v8i aU1 = lds_a_frag(sAu[buf], rowW + 16, ks, lane);
            v8i bG0 = lds_b_frag(sBg[buf], colW,      ks, lane);
            v8i bG1 = lds_b_frag(sBg[buf], colW + 16, ks, lane);
            v8i bU0 = lds_b_frag(sBu[buf], colW,      ks, lane);
            v8i bU1 = lds_b_frag(sBu[buf], colW + 16, ks, lane);

            accG[0][0] = wmma_iu8(aG0, bG0, accG[0][0]);
            accG[0][1] = wmma_iu8(aG0, bG1, accG[0][1]);
            accG[1][0] = wmma_iu8(aG1, bG0, accG[1][0]);
            accG[1][1] = wmma_iu8(aG1, bG1, accG[1][1]);
            accU[0][0] = wmma_iu8(aU0, bU0, accU[0][0]);
            accU[0][1] = wmma_iu8(aU0, bU1, accU[0][1]);
            accU[1][0] = wmma_iu8(aU1, bU0, accU[1][0]);
            accU[1][1] = wmma_iu8(aU1, bU1, accU[1][1]);
        }
    }

    const int moff = (lane >> 4) << 3;
    const int nloc = lane & 15;
#pragma unroll
    for (int mi = 0; mi < 2; ++mi)
#pragma unroll
        for (int ni = 0; ni < 2; ++ni)
#pragma unroll
            for (int r = 0; r < 8; ++r) {
                int row = rowB + rowW + mi * 16 + moff + r;
                int col = colB + colW + ni * 16 + nloc;
                float g = (float)accG[mi][ni][r] * sg[row] * swg[col];
                float u = (float)accU[mi][ni][r] * su[row] * swu[col];
                float v = (g / (1.0f + __expf(-g))) * u;
                v = fminf(fmaxf(v, -SWIGLU_LIMIT), SWIGLU_LIMIT);
                inter[(size_t)row * I_DIM + col] = __float2bfloat16(v);
            }
}

__global__ __launch_bounds__(256) void quant_inter_kernel(
    const __hip_bfloat16* __restrict__ inter, const float* __restrict__ inv_i,
    signed char* __restrict__ qi, float* __restrict__ si)
{
    __shared__ float red[256];
    const int t = blockIdx.x;
    const __hip_bfloat16* xr = inter + (size_t)t * I_DIM;

    float vv[8];
    float m = 0.0f;
#pragma unroll
    for (int i = 0; i < 8; ++i) {
        int d = threadIdx.x + i * 256;
        float v = __bfloat162float(xr[d]) * inv_i[d];
        vv[i] = v;
        m = fmaxf(m, fabsf(v));
    }
    red[threadIdx.x] = m;
    __syncthreads();
    for (int s = 128; s > 0; s >>= 1) {
        if (threadIdx.x < s)
            red[threadIdx.x] = fmaxf(red[threadIdx.x], red[threadIdx.x + s]);
        __syncthreads();
    }
    float sc = fmaxf(red[0] * (1.0f / QMAXF), 1e-8f);
    if (threadIdx.x == 0) si[t] = sc;
    float rs = 1.0f / sc;
#pragma unroll
    for (int i = 0; i < 8; ++i) {
        int d = threadIdx.x + i * 256;
        qi[(size_t)t * I_DIM + d] =
            (signed char)(int)fminf(fmaxf(rintf(vv[i] * rs), -QMAXF), QMAXF);
    }
}

__global__ __launch_bounds__(256) void gemm_down_kernel(
    const signed char* __restrict__ qi, const float* __restrict__ si,
    const signed char* __restrict__ wd, const float* __restrict__ swd,
    float* __restrict__ out)
{
    __shared__ signed char sA[2][64 * LDS_STRIDE];
    __shared__ signed char sB[2][128 * LDS_STRIDE];

    const int tid  = threadIdx.x;
    const int lane = tid & 31;
    const int wave = tid >> 5;
    const int rowB = blockIdx.y * 64;
    const int colB = blockIdx.x * 128;
    const int rowW = (wave & 1) * 32;
    const int colW = (wave >> 1) * 32;

    auto stage = [&](int buf, int k0) {
#pragma unroll
        for (int j = 0; j < 2; ++j) {
            int c = tid + (j << 8);
            int r = c >> 3, o = (c & 7) << 4;
            async_b128(lds_off32(&sA[buf][r * LDS_STRIDE + o]),
                       qi + (size_t)(rowB + r) * I_DIM + k0 + o);
        }
#pragma unroll
        for (int j = 0; j < 4; ++j) {
            int c = tid + (j << 8);
            int r = c >> 3, o = (c & 7) << 4;
            async_b128(lds_off32(&sB[buf][r * LDS_STRIDE + o]),
                       wd + (size_t)(colB + r) * I_DIM + k0 + o);
        }
    };

    v8i acc[2][2];
#pragma unroll
    for (int mi = 0; mi < 2; ++mi)
#pragma unroll
        for (int ni = 0; ni < 2; ++ni)
            acc[mi][ni] = (v8i)(0);

    stage(0, 0);
    for (int k0 = 0; k0 < I_DIM; k0 += KSTEP) {
        const int buf = (k0 >> 7) & 1;
        __syncthreads();
        if (k0 + KSTEP < I_DIM) {
            stage(buf ^ 1, k0 + KSTEP);
            asm volatile("s_wait_asynccnt 0x6" ::: "memory");
        } else {
            asm volatile("s_wait_asynccnt 0x0" ::: "memory");
        }
        __syncthreads();

#pragma unroll
        for (int ks = 0; ks < KSTEP; ks += 64) {
            v8i a0 = lds_a_frag(sA[buf], rowW,      ks, lane);
            v8i a1 = lds_a_frag(sA[buf], rowW + 16, ks, lane);
            v8i b0 = lds_b_frag(sB[buf], colW,      ks, lane);
            v8i b1 = lds_b_frag(sB[buf], colW + 16, ks, lane);

            acc[0][0] = wmma_iu8(a0, b0, acc[0][0]);
            acc[0][1] = wmma_iu8(a0, b1, acc[0][1]);
            acc[1][0] = wmma_iu8(a1, b0, acc[1][0]);
            acc[1][1] = wmma_iu8(a1, b1, acc[1][1]);
        }
    }

    const int moff = (lane >> 4) << 3;
    const int nloc = lane & 15;
#pragma unroll
    for (int mi = 0; mi < 2; ++mi)
#pragma unroll
        for (int ni = 0; ni < 2; ++ni)
#pragma unroll
            for (int r = 0; r < 8; ++r) {
                int row = rowB + rowW + mi * 16 + moff + r;
                int col = colB + colW + ni * 16 + nloc;
                out[(size_t)row * H_DIM + col] =
                    (float)acc[mi][ni][r] * si[row] * swd[col];
            }
}

extern "C" void kernel_launch(void* const* d_in, const int* in_sizes, int n_in,
                              void* d_out, int out_size, void* d_ws, size_t ws_size,
                              hipStream_t stream) {
    const float* x        = (const float*)d_in[0];
    const float* w_gate   = (const float*)d_in[1];
    const float* s_wgate  = (const float*)d_in[2];
    const float* w_up     = (const float*)d_in[3];
    const float* s_wup    = (const float*)d_in[4];
    const float* w_down   = (const float*)d_in[5];
    const float* s_wdown  = (const float*)d_in[6];
    const float* inv_gate = (const float*)d_in[7];
    const float* inv_up   = (const float*)d_in[8];
    const float* inv_int  = (const float*)d_in[9];

    const size_t T = (size_t)in_sizes[0] / H_DIM;

    char* ws = (char*)d_ws;
    size_t off = 0;
    auto carve = [&](size_t bytes) -> char* {
        char* p = ws + off;
        off = (off + bytes + 255) & ~(size_t)255;
        return p;
    };
    signed char* qg  = (signed char*)carve(T * H_DIM);
    signed char* qu  = (signed char*)carve(T * H_DIM);
    signed char* wqg = (signed char*)carve((size_t)I_DIM * H_DIM);
    signed char* wqu = (signed char*)carve((size_t)I_DIM * H_DIM);
    signed char* wqd = (signed char*)carve((size_t)H_DIM * I_DIM);
    signed char* qi  = (signed char*)carve(T * I_DIM);
    __hip_bfloat16* inter = (__hip_bfloat16*)carve(T * I_DIM * sizeof(__hip_bfloat16));
    float* sg = (float*)carve(T * sizeof(float));
    float* su = (float*)carve(T * sizeof(float));
    float* si = (float*)carve(T * sizeof(float));
    (void)ws_size; (void)n_in; (void)out_size;

    quant_x_kernel<<<dim3((unsigned)T), 256, 0, stream>>>(
        x, inv_gate, inv_up, qg, qu, sg, su);

    const int nwg = I_DIM * H_DIM;
    convert_w_kernel<<<dim3((nwg + 255) / 256), 256, 0, stream>>>(w_gate, wqg, nwg);
    convert_w_kernel<<<dim3((nwg + 255) / 256), 256, 0, stream>>>(w_up,   wqu, nwg);
    convert_w_kernel<<<dim3((nwg + 255) / 256), 256, 0, stream>>>(w_down, wqd, nwg);

    gemm_gate_up_kernel<<<dim3(I_DIM / 128, (unsigned)(T / 64)), 256, 0, stream>>>(
        qg, qu, sg, su, wqg, wqu, s_wgate, s_wup, inter);

    quant_inter_kernel<<<dim3((unsigned)T), 256, 0, stream>>>(inter, inv_int, qi, si);

    gemm_down_kernel<<<dim3(H_DIM / 128, (unsigned)(T / 64)), 256, 0, stream>>>(
        qi, si, wqd, s_wdown, (float*)d_out);
}
